// SCOAD_1460288880931
// MI455X (gfx1250) — matrix operand order repaired and checked by reading, not compiled, not run on hardware
//
#include <hip/hip_runtime.h>
#include <hip/hip_bf16.h>
#include <stdint.h>

#define DI __device__ __forceinline__

typedef __attribute__((ext_vector_type(16))) __bf16 v16bf;
typedef __attribute__((ext_vector_type(8)))  float  v8f;
typedef __attribute__((ext_vector_type(8)))  __bf16 bf16x8;
typedef __attribute__((ext_vector_type(4)))  unsigned int u32x4;
typedef __attribute__((ext_vector_type(4)))  int i32x4;
typedef __attribute__((ext_vector_type(8)))  int i32x8;

#define BB    16
#define TT    512
#define FEATD 2048
#define HIDD  1024
#define NCLSD 21
#define STEPL 64
#define MROWS (BB * TT)      /* 8192 */
#define NCHUNK 128           /* B * (T/STEP) */

// ---------------------------------------------------------------------------
// Tensor Data Mover: load a 2D tile (tileRows x tileCols bf16 elements, row
// stride rowStrideElems) from global into LDS at byte offset ldsOff.
// Descriptor packing per CDNA5 ISA ch.8 (D# group0/group1), 2D tensor,
// data_size=1 (2 bytes), no gather/iterate/pad, no multicast.
// ---------------------------------------------------------------------------
DI unsigned lds_byte_off(const void* p) {
    return (unsigned)(unsigned long long)(uintptr_t)p;
}

DI void tdm_load_2d(const void* gptr, unsigned ldsOff, int tileCols, int tileRows,
                    long long rowStrideElems, int tensorRows) {
    unsigned long long ga = (unsigned long long)(uintptr_t)gptr;
    unsigned td0 = (unsigned)rowStrideElems;        // tensor_dim0 (elements)
    unsigned td1 = (unsigned)tensorRows;            // tensor_dim1 (rows)
    unsigned long long s0 = (unsigned long long)rowStrideElems;

    u32x4 g0 = {0u, 0u, 0u, 0u};
    g0[0] = 1u;                                          // count=1, user mode
    g0[1] = ldsOff;                                      // lds_addr
    g0[2] = (unsigned)(ga & 0xffffffffull);              // global_addr[31:0]
    g0[3] = (unsigned)((ga >> 32) & 0x1ffffffull)        // global_addr[56:32]
          | (2u << 30);                                  // type=2 ("image")

    i32x8 g1 = {0, 0, 0, 0, 0, 0, 0, 0};
    g1[0] = (int)(1u << 16);                             // data_size=1 (2B)
    g1[1] = (int)((td0 & 0xffffu) << 16);                // tensor_dim0[15:0]
    g1[2] = (int)(((td0 >> 16) & 0xffffu) | ((td1 & 0xffffu) << 16));
    g1[3] = (int)(((td1 >> 16) & 0xffffu) | (((unsigned)tileCols & 0xffffu) << 16));
    g1[4] = (int)((unsigned)tileRows & 0xffffu);         // tile_dim1
    g1[5] = (int)(s0 & 0xffffffffull);                   // tensor_dim0_stride lo
    g1[6] = (int)((s0 >> 32) & 0xffffull);               // stride hi, dim1_stride=0
    g1[7] = 0;

    i32x4 z4 = {0, 0, 0, 0};
#if __clang_major__ >= 23
    i32x8 z8 = {0, 0, 0, 0, 0, 0, 0, 0};
    __builtin_amdgcn_tensor_load_to_lds(g0, g1, z4, z4, z8, 0);
#else
    __builtin_amdgcn_tensor_load_to_lds(g0, g1, z4, z4, 0);
#endif
}

// ---------------------------------------------------------------------------
// WMMA fragment loaders (bf16, 16x16x32). ISA layouts:
//  A 16x32: lane m (0..15): K 0..7 then 16..23 ; lane m+16: K 8..15 then 24..31
//  B 32x16: lane n (0..15): K 0..7 then 16..23 ; lane n+16: K 8..15 then 24..31
// rowp points at the start of a 32-element (64B, 16B-aligned) LDS row.
// ---------------------------------------------------------------------------
DI v16bf load_a_frag(const __bf16* rowp, int hl) {
    bf16x8 lo = *(const bf16x8*)(rowp + hl * 8);
    bf16x8 hi = *(const bf16x8*)(rowp + 16 + hl * 8);
    v16bf a;
#pragma unroll
    for (int e = 0; e < 8; ++e) { a[e] = lo[e]; a[e + 8] = hi[e]; }
    return a;
}

DI v16bf load_b_frag(const __bf16* rowp, int hl) {
    bf16x8 lo = *(const bf16x8*)(rowp + hl * 8);
    bf16x8 hi = *(const bf16x8*)(rowp + 16 + hl * 8);
    v16bf b;
#pragma unroll
    for (int e = 0; e < 8; ++e) { b[e] = lo[e]; b[e + 8] = hi[e]; }
    return b;
}

DI v8f wmma_bf16(v16bf a, v16bf b, v8f c) {
    return __builtin_amdgcn_wmma_f32_16x16x32_bf16(false, a, false, b,
                                                   (short)0, c, false, false);
}

DI float sigmoidf(float x) { return 1.0f / (1.0f + __expf(-x)); }

// murmur3 fmix32: deterministic dropout mask hash
DI float dropout(float v, unsigned idx) {
    unsigned h = idx;
    h ^= h >> 16; h *= 0x85ebca6bu; h ^= h >> 13; h *= 0xc2b2ae35u; h ^= h >> 16;
    return ((h & 0xffffffu) < 11744051u) ? v * 1.42857143f : 0.0f;  // keep=0.7
}

// ---------------------------------------------------------------------------
// fc: X = dropout(relu(feats @ fc_w^T + b)). M=8192, N=K=2048.
// Block 256 thr (8 waves), tile 128x64, wave tile 32x32 (4 wmma / K-chunk).
// ---------------------------------------------------------------------------
__global__ __launch_bounds__(256) void fc_gemm_kernel(
    const __bf16* __restrict__ A, const __bf16* __restrict__ W,
    const float* __restrict__ bias, float* __restrict__ Xout,
    __bf16* __restrict__ Xbf) {
    const int N = FEATD, K = FEATD;
    __shared__ __align__(16) __bf16 As[128][32];
    __shared__ __align__(16) __bf16 Ws[64][32];
    const int tid = threadIdx.x;
    const int lane = tid & 31, wv = tid >> 5;
    const int hl = lane >> 4, l16 = lane & 15;
    const int bm = blockIdx.y * 128;
    const int bn = blockIdx.x * 64;
    const int wm = (wv >> 1) * 32;
    const int wn = (wv & 1) * 32;

    v8f acc[2][2];
#pragma unroll
    for (int i = 0; i < 2; ++i)
#pragma unroll
        for (int j = 0; j < 2; ++j) acc[i][j] = (v8f){0,0,0,0,0,0,0,0};

    for (int k0 = 0; k0 < K; k0 += 32) {
#pragma unroll
        for (int it = 0; it < 2; ++it) {
            int c = tid + it * 256;
            int row = c >> 2, seg = c & 3;
            *(bf16x8*)&As[row][seg * 8] =
                *(const bf16x8*)(A + (size_t)(bm + row) * K + k0 + seg * 8);
        }
        if (wv == 0) {
            tdm_load_2d(W + (size_t)bn * K + k0, lds_byte_off(&Ws[0][0]),
                        32, 64, K, N);
            __builtin_amdgcn_s_wait_tensorcnt((short)0);
        }
        __syncthreads();
        v16bf af[2], bfr[2];
        af[0]  = load_a_frag(&As[wm + l16][0], hl);
        af[1]  = load_a_frag(&As[wm + 16 + l16][0], hl);
        bfr[0] = load_b_frag(&Ws[wn + l16][0], hl);
        bfr[1] = load_b_frag(&Ws[wn + 16 + l16][0], hl);
#pragma unroll
        for (int sm = 0; sm < 2; ++sm)
#pragma unroll
            for (int sn = 0; sn < 2; ++sn)
                acc[sm][sn] = wmma_bf16(af[sm], bfr[sn], acc[sm][sn]);
        __syncthreads();
    }

#pragma unroll
    for (int sm = 0; sm < 2; ++sm)
#pragma unroll
        for (int sn = 0; sn < 2; ++sn) {
            int col = bn + wn + sn * 16 + l16;
            float bv = bias[col];
#pragma unroll
            for (int r = 0; r < 8; ++r) {
                int row = bm + wm + sm * 16 + hl * 8 + r;
                float v = fmaxf(acc[sm][sn][r] + bv, 0.0f);
                unsigned idx = (unsigned)row * (unsigned)N + (unsigned)col;
                v = dropout(v, idx);
                size_t o = (size_t)row * N + col;
                Xout[o] = v;
                Xbf[o] = (__bf16)v;
            }
        }
}

// ---------------------------------------------------------------------------
// classifier: vl = X @ cls_w^T + b. M=8192, N=32(pad of 21), K=2048.
// Block tile 128x32, wave = 16 rows x 32 cols (2 wmma / K-chunk).
// ---------------------------------------------------------------------------
__global__ __launch_bounds__(256) void cls_gemm_kernel(
    const __bf16* __restrict__ A, const __bf16* __restrict__ W,
    const float* __restrict__ bias, float* __restrict__ out) {
    const int K = FEATD;
    __shared__ __align__(16) __bf16 As[128][32];
    __shared__ __align__(16) __bf16 Ws[32][32];
    const int tid = threadIdx.x;
    const int lane = tid & 31, wv = tid >> 5;
    const int hl = lane >> 4, l16 = lane & 15;
    const int bm = blockIdx.x * 128;

    v8f acc[2] = {(v8f){0,0,0,0,0,0,0,0}, (v8f){0,0,0,0,0,0,0,0}};

    for (int k0 = 0; k0 < K; k0 += 32) {
#pragma unroll
        for (int it = 0; it < 2; ++it) {
            int c = tid + it * 256;
            int row = c >> 2, seg = c & 3;
            *(bf16x8*)&As[row][seg * 8] =
                *(const bf16x8*)(A + (size_t)(bm + row) * K + k0 + seg * 8);
        }
        if (wv == 0) {
            tdm_load_2d(W + k0, lds_byte_off(&Ws[0][0]), 32, 32, K, 32);
            __builtin_amdgcn_s_wait_tensorcnt((short)0);
        }
        __syncthreads();
        v16bf a  = load_a_frag(&As[wv * 16 + l16][0], hl);
        v16bf b0 = load_b_frag(&Ws[l16][0], hl);
        v16bf b1 = load_b_frag(&Ws[16 + l16][0], hl);
        acc[0] = wmma_bf16(a, b0, acc[0]);
        acc[1] = wmma_bf16(a, b1, acc[1]);
        __syncthreads();
    }

#pragma unroll
    for (int sn = 0; sn < 2; ++sn) {
        int col = sn * 16 + l16;
        if (col < NCLSD) {
            float bv = bias[col];
#pragma unroll
            for (int r = 0; r < 8; ++r) {
                int row = bm + wv * 16 + hl * 8 + r;
                out[(size_t)row * NCLSD + col] = acc[sn][r] + bv;
            }
        }
    }
}

// ---------------------------------------------------------------------------
// GRU gates: y==0 -> gi = x_t @ w_ih^T + b_ih (K=2048)
//            y==1 -> gh = h   @ w_hh^T + b_hh (K=1024)
// M=128 (all chunks), N=3072, block tile 128x64.
// x_t rows: global row = chunk*STEP + t.
// ---------------------------------------------------------------------------
__global__ __launch_bounds__(256) void gates_gemm_kernel(
    const __bf16* __restrict__ xbf, const __bf16* __restrict__ hbf,
    const __bf16* __restrict__ wih, const __bf16* __restrict__ whh,
    const float* __restrict__ bih, const float* __restrict__ bhh,
    float* __restrict__ gi, float* __restrict__ gh, int t) {
    const bool isH = (blockIdx.y == 1);
    const int K = isH ? HIDD : FEATD;
    const __bf16* W = isH ? whh : wih;
    const float* bias = isH ? bhh : bih;
    float* out = isH ? gh : gi;

    __shared__ __align__(16) __bf16 As[128][32];
    __shared__ __align__(16) __bf16 Ws[64][32];
    const int tid = threadIdx.x;
    const int lane = tid & 31, wv = tid >> 5;
    const int hl = lane >> 4, l16 = lane & 15;
    const int bn = blockIdx.x * 64;
    const int wm = (wv >> 1) * 32;
    const int wn = (wv & 1) * 32;

    v8f acc[2][2];
#pragma unroll
    for (int i = 0; i < 2; ++i)
#pragma unroll
        for (int j = 0; j < 2; ++j) acc[i][j] = (v8f){0,0,0,0,0,0,0,0};

    for (int k0 = 0; k0 < K; k0 += 32) {
#pragma unroll
        for (int it = 0; it < 2; ++it) {
            int c = tid + it * 256;
            int row = c >> 2, seg = c & 3;
            const __bf16* src = isH
                ? (hbf + (size_t)row * HIDD + k0 + seg * 8)
                : (xbf + ((size_t)row * STEPL + t) * FEATD + k0 + seg * 8);
            *(bf16x8*)&As[row][seg * 8] = *(const bf16x8*)src;
        }
        if (wv == 0) {
            tdm_load_2d(W + (size_t)bn * K + k0, lds_byte_off(&Ws[0][0]),
                        32, 64, K, 3 * HIDD);
            __builtin_amdgcn_s_wait_tensorcnt((short)0);
        }
        __syncthreads();
        v16bf af[2], bfr[2];
        af[0]  = load_a_frag(&As[wm + l16][0], hl);
        af[1]  = load_a_frag(&As[wm + 16 + l16][0], hl);
        bfr[0] = load_b_frag(&Ws[wn + l16][0], hl);
        bfr[1] = load_b_frag(&Ws[wn + 16 + l16][0], hl);
#pragma unroll
        for (int sm = 0; sm < 2; ++sm)
#pragma unroll
            for (int sn = 0; sn < 2; ++sn)
                acc[sm][sn] = wmma_bf16(af[sm], bfr[sn], acc[sm][sn]);
        __syncthreads();
    }

#pragma unroll
    for (int sm = 0; sm < 2; ++sm)
#pragma unroll
        for (int sn = 0; sn < 2; ++sn) {
            int col = bn + wn + sn * 16 + l16;
            float bv = bias[col];
#pragma unroll
            for (int r = 0; r < 8; ++r) {
                int row = wm + sm * 16 + hl * 8 + r;
                out[(size_t)row * (3 * HIDD) + col] = acc[sm][sn][r] + bv;
            }
        }
}

// ---------------------------------------------------------------------------
// GRU elementwise update (torch gate order r,z,n).
// ---------------------------------------------------------------------------
__global__ __launch_bounds__(256) void gru_update_kernel(
    const float* __restrict__ gi, const float* __restrict__ gh,
    float* __restrict__ h, __bf16* __restrict__ hbf, __bf16* __restrict__ hrbf) {
    int idx = blockIdx.x * 256 + threadIdx.x;   // < 128*1024
    int m = idx >> 10, j = idx & (HIDD - 1);
    size_t base = (size_t)m * (3 * HIDD);
    float r = sigmoidf(gi[base + j] + gh[base + j]);
    float z = sigmoidf(gi[base + HIDD + j] + gh[base + HIDD + j]);
    float n = tanhf(gi[base + 2 * HIDD + j] + r * gh[base + 2 * HIDD + j]);
    float hv = (1.0f - z) * n + z * h[idx];
    h[idx] = hv;
    hbf[idx] = (__bf16)hv;
    hrbf[idx] = (__bf16)fmaxf(hv, 0.0f);
}

// ---------------------------------------------------------------------------
// heads: [enc | mask] = relu(h) @ [cf_w ; mk_w]^T. M=128, N=32(22+2+pad), K=1024.
// One block; scatter cols 0..21 -> enc, 22..23 -> mask at step t.
// ---------------------------------------------------------------------------
__global__ __launch_bounds__(256) void head_gemm_kernel(
    const __bf16* __restrict__ A, const __bf16* __restrict__ W,
    const float* __restrict__ cfb, const float* __restrict__ mkb,
    float* __restrict__ enc, float* __restrict__ msk, int t) {
    const int K = HIDD;
    __shared__ __align__(16) __bf16 As[128][32];
    __shared__ __align__(16) __bf16 Ws[32][32];
    const int tid = threadIdx.x;
    const int lane = tid & 31, wv = tid >> 5;
    const int hl = lane >> 4, l16 = lane & 15;

    v8f acc[2] = {(v8f){0,0,0,0,0,0,0,0}, (v8f){0,0,0,0,0,0,0,0}};

    for (int k0 = 0; k0 < K; k0 += 32) {
#pragma unroll
        for (int it = 0; it < 2; ++it) {
            int c = tid + it * 256;
            int row = c >> 2, seg = c & 3;
            *(bf16x8*)&As[row][seg * 8] =
                *(const bf16x8*)(A + (size_t)row * K + k0 + seg * 8);
        }
        if (wv == 0) {
            tdm_load_2d(W + k0, lds_byte_off(&Ws[0][0]), 32, 32, K, 32);
            __builtin_amdgcn_s_wait_tensorcnt((short)0);
        }
        __syncthreads();
        v16bf a  = load_a_frag(&As[wv * 16 + l16][0], hl);
        v16bf b0 = load_b_frag(&Ws[l16][0], hl);
        v16bf b1 = load_b_frag(&Ws[16 + l16][0], hl);
        acc[0] = wmma_bf16(a, b0, acc[0]);
        acc[1] = wmma_bf16(a, b1, acc[1]);
        __syncthreads();
    }

#pragma unroll
    for (int sn = 0; sn < 2; ++sn) {
        int col = sn * 16 + l16;
#pragma unroll
        for (int r = 0; r < 8; ++r) {
            int row = wv * 16 + hl * 8 + r;
            size_t gr = (size_t)t * NCHUNK + row;
            float v = acc[sn][r];
            if (col < NCLSD + 1)
                enc[gr * (NCLSD + 1) + col] = v + cfb[col];
            else if (col < NCLSD + 3)
                msk[gr * 2 + (col - (NCLSD + 1))] = v + mkb[col - (NCLSD + 1)];
        }
    }
}

// ---------------------------------------------------------------------------
// prep kernels
// ---------------------------------------------------------------------------
__global__ void cvt_f32_bf16_kernel(const float* __restrict__ in,
                                    __bf16* __restrict__ out, int n) {
    int i = blockIdx.x * blockDim.x + threadIdx.x;
    int stride = gridDim.x * blockDim.x;
    for (; i < n; i += stride) out[i] = (__bf16)in[i];
}

__global__ void pack_cls_kernel(const float* __restrict__ w,
                                __bf16* __restrict__ out) {
    int i = blockIdx.x * blockDim.x + threadIdx.x;   // 32*2048
    int r = i >> 11, c = i & (FEATD - 1);
    out[i] = (r < NCLSD) ? (__bf16)w[r * FEATD + c] : (__bf16)0.0f;
}

__global__ void pack_head_kernel(const float* __restrict__ cf,
                                 const float* __restrict__ mk,
                                 __bf16* __restrict__ out) {
    int i = blockIdx.x * blockDim.x + threadIdx.x;   // 32*1024
    int r = i >> 10, c = i & (HIDD - 1);
    float v = 0.0f;
    if (r < NCLSD + 1) v = cf[r * HIDD + c];
    else if (r < NCLSD + 3) v = mk[(r - (NCLSD + 1)) * HIDD + c];
    out[i] = (__bf16)v;
}

__global__ void zero_h_kernel(float* __restrict__ h, __bf16* __restrict__ hbf) {
    int i = blockIdx.x * 256 + threadIdx.x;          // 128*1024
    h[i] = 0.0f;
    hbf[i] = (__bf16)0.0f;
}

// ---------------------------------------------------------------------------
extern "C" void kernel_launch(void* const* d_in, const int* in_sizes, int n_in,
                              void* d_out, int out_size, void* d_ws, size_t ws_size,
                              hipStream_t stream) {
    const float* feats = (const float*)d_in[0];
    const float* fc_w  = (const float*)d_in[1];
    const float* fc_b  = (const float*)d_in[2];
    const float* cls_w = (const float*)d_in[3];
    const float* cls_b = (const float*)d_in[4];
    const float* w_ih  = (const float*)d_in[5];
    const float* w_hh  = (const float*)d_in[6];
    const float* b_ih  = (const float*)d_in[7];
    const float* b_hh  = (const float*)d_in[8];
    const float* cf_w  = (const float*)d_in[9];
    const float* cf_b  = (const float*)d_in[10];
    const float* mk_w  = (const float*)d_in[11];
    const float* mk_b  = (const float*)d_in[12];

    float* out  = (float*)d_out;
    float* enc  = out;                                   // 8192*22
    float* msk  = out + 8192 * 22;                       // 8192*2
    float* xout = out + 8192 * 22 + 8192 * 2;            // 8192*2048
    float* vl   = xout + (size_t)MROWS * FEATD;          // 8192*21

    char* ws = (char*)d_ws;
    size_t off = 0;
    auto alloc = [&](size_t bytes) -> void* {
        void* p = ws + off;
        off += (bytes + 255) & ~(size_t)255;
        return p;
    };
    __bf16* feats_bf  = (__bf16*)alloc((size_t)MROWS * FEATD * 2);
    __bf16* fc_w_bf   = (__bf16*)alloc((size_t)FEATD * FEATD * 2);
    __bf16* x_bf      = (__bf16*)alloc((size_t)MROWS * FEATD * 2);
    __bf16* cls_w_bf  = (__bf16*)alloc((size_t)32 * FEATD * 2);
    __bf16* w_ih_bf   = (__bf16*)alloc((size_t)3 * HIDD * FEATD * 2);
    __bf16* w_hh_bf   = (__bf16*)alloc((size_t)3 * HIDD * HIDD * 2);
    __bf16* head_w_bf = (__bf16*)alloc((size_t)32 * HIDD * 2);
    float*  h     = (float*)alloc((size_t)NCHUNK * HIDD * 4);
    __bf16* h_bf  = (__bf16*)alloc((size_t)NCHUNK * HIDD * 2);
    __bf16* hr_bf = (__bf16*)alloc((size_t)NCHUNK * HIDD * 2);
    float*  gi    = (float*)alloc((size_t)NCHUNK * 3 * HIDD * 4);
    float*  gh    = (float*)alloc((size_t)NCHUNK * 3 * HIDD * 4);

    cvt_f32_bf16_kernel<<<2048, 256, 0, stream>>>(feats, feats_bf, MROWS * FEATD);
    cvt_f32_bf16_kernel<<<2048, 256, 0, stream>>>(fc_w, fc_w_bf, FEATD * FEATD);
    cvt_f32_bf16_kernel<<<2048, 256, 0, stream>>>(w_ih, w_ih_bf, 3 * HIDD * FEATD);
    cvt_f32_bf16_kernel<<<2048, 256, 0, stream>>>(w_hh, w_hh_bf, 3 * HIDD * HIDD);
    pack_cls_kernel<<<(32 * FEATD) / 256, 256, 0, stream>>>(cls_w, cls_w_bf);
    pack_head_kernel<<<(32 * HIDD) / 256, 256, 0, stream>>>(cf_w, mk_w, head_w_bf);
    zero_h_kernel<<<(NCHUNK * HIDD) / 256, 256, 0, stream>>>(h, h_bf);

    fc_gemm_kernel<<<dim3(FEATD / 64, MROWS / 128), 256, 0, stream>>>(
        feats_bf, fc_w_bf, fc_b, xout, x_bf);
    cls_gemm_kernel<<<MROWS / 128, 256, 0, stream>>>(x_bf, cls_w_bf, cls_b, vl);

    for (int t = 0; t < STEPL; ++t) {
        gates_gemm_kernel<<<dim3((3 * HIDD) / 64, 2), 256, 0, stream>>>(
            x_bf, h_bf, w_ih_bf, w_hh_bf, b_ih, b_hh, gi, gh, t);
        gru_update_kernel<<<(NCHUNK * HIDD) / 256, 256, 0, stream>>>(
            gi, gh, h, h_bf, hr_bf);
        head_gemm_kernel<<<1, 256, 0, stream>>>(hr_bf, head_w_bf, cf_b, mk_b,
                                                enc, msk, t);
    }
    (void)in_sizes; (void)n_in; (void)out_size; (void)ws_size;
}
